// PFGT_46849503265073
// MI455X (gfx1250) — matrix operand order repaired and checked by reading, not compiled, not run on hardware
//
#include <hip/hip_runtime.h>
#include <hip/hip_bf16.h>

#define NNODES 100000
#define NEDGES 800000
#define F_IN   128
#define C_HID  32
#define N_CLS  16
#define K_HOPS 3
#define CSTF   1e-5f

typedef _Float16 h16 __attribute__((ext_vector_type(16)));
typedef _Float16 h8  __attribute__((ext_vector_type(8)));
typedef float    f8  __attribute__((ext_vector_type(8)));

// ---------------------------------------------------------------------------
// Kernel 1: fused node transform with WMMA (wave32, 16 nodes per wave)
//   x = relu(feat @ Win + bin)      [16 x 32]  via 4 K-chunks x 2 N-tiles
//   Q = 1+elu(x@Wq+bq), K = 1+elu(x@Wk+bk), V = x@Wv+bv
// ---------------------------------------------------------------------------
__global__ __launch_bounds__(256)
void node_transform(const float* __restrict__ feat,
                    const float* __restrict__ Win, const float* __restrict__ bin,
                    const float* __restrict__ Wq,  const float* __restrict__ bq,
                    const float* __restrict__ Wk,  const float* __restrict__ bk,
                    const float* __restrict__ Wv,  const float* __restrict__ bv,
                    float* __restrict__ Qo, float* __restrict__ Ko,
                    float* __restrict__ Vo)
{
    __shared__ __align__(16) _Float16 lds_x[8][16 * C_HID];  // per-wave x tile (f16)

    const int wave = threadIdx.x >> 5;
    const int lane = threadIdx.x & 31;
    const int tile = blockIdx.x * 8 + wave;
    if (tile >= NNODES / 16) return;
    const int  node0 = tile * 16;
    const int  m     = lane & 15;
    const bool lo    = lane < 16;

    // -------- x = relu(feat @ Win + bin) --------
    f8 acc[2];
#pragma unroll
    for (int i = 0; i < 8; ++i) { acc[0][i] = 0.f; acc[1][i] = 0.f; }

    for (int kc = 0; kc < F_IN / 32; ++kc) {
        // A fragment: 16x32 f16 tile of node features.
        // lanes 0-15: halves 0..15 -> K=0..15
        // lanes16-31: halves 0..15 -> K=16..31
        h16 a;
        const float* rp = feat + (size_t)(node0 + m) * F_IN + kc * 32 + (lo ? 0 : 16);
        const float4 a0 = *(const float4*)(rp);
        const float4 a1 = *(const float4*)(rp + 4);
        const float4 a2 = *(const float4*)(rp + 8);
        const float4 a3 = *(const float4*)(rp + 12);
        a[0]=(_Float16)a0.x; a[1]=(_Float16)a0.y; a[2]=(_Float16)a0.z; a[3]=(_Float16)a0.w;
        a[4]=(_Float16)a1.x; a[5]=(_Float16)a1.y; a[6]=(_Float16)a1.z; a[7]=(_Float16)a1.w;
        a[8]=(_Float16)a2.x; a[9]=(_Float16)a2.y; a[10]=(_Float16)a2.z; a[11]=(_Float16)a2.w;
        a[12]=(_Float16)a3.x; a[13]=(_Float16)a3.y; a[14]=(_Float16)a3.z; a[15]=(_Float16)a3.w;

#pragma unroll
        for (int nt = 0; nt < 2; ++nt) {
            // B fragment: 32x16 f16; lanes 0-15 hold K=0..15, lanes 16-31 K=16..31
            h16 b;
            const int N  = nt * 16 + m;
            const int kb = kc * 32 + (lo ? 0 : 16);
#pragma unroll
            for (int h = 0; h < 16; ++h)
                b[h] = (_Float16)Win[(kb + h) * C_HID + N];
            acc[nt] = __builtin_amdgcn_wmma_f32_16x16x32_f16(
                false, a, false, b, (short)0, acc[nt], false, false);
        }
    }

    // bias + relu -> LDS (f16), C-layout -> row-major
#pragma unroll
    for (int nt = 0; nt < 2; ++nt) {
        const int   col = nt * 16 + m;
        const float bb  = bin[col];
#pragma unroll
        for (int r = 0; r < 8; ++r) {
            const int row = r + (lo ? 0 : 8);
            float v = acc[nt][r] + bb;
            v = v > 0.f ? v : 0.f;
            lds_x[wave][row * C_HID + col] = (_Float16)v;
        }
    }

    // reload x as A fragment (single K-chunk of 32); same-wave DS ordering
    h16 xa;
    {
        const _Float16* xp = &lds_x[wave][m * C_HID + (lo ? 0 : 16)];
        const h8 x0 = *(const h8*)(xp);
        const h8 x1 = *(const h8*)(xp + 8);
#pragma unroll
        for (int h = 0; h < 8; ++h) { xa[h] = x0[h]; xa[h + 8] = x1[h]; }
    }

    // -------- Q = 1+elu(x@Wq+bq), K = 1+elu(x@Wk+bk) --------
#pragma unroll
    for (int nt = 0; nt < 2; ++nt) {
        const int N  = nt * 16 + m;
        const int kb = lo ? 0 : 16;
        h16 bqf, bkf;
#pragma unroll
        for (int h = 0; h < 16; ++h) {
            bqf[h] = (_Float16)Wq[(kb + h) * C_HID + N];
            bkf[h] = (_Float16)Wk[(kb + h) * C_HID + N];
        }
        f8 cq, ck;
#pragma unroll
        for (int i = 0; i < 8; ++i) { cq[i] = 0.f; ck[i] = 0.f; }
        cq = __builtin_amdgcn_wmma_f32_16x16x32_f16(false, xa, false, bqf, (short)0, cq, false, false);
        ck = __builtin_amdgcn_wmma_f32_16x16x32_f16(false, xa, false, bkf, (short)0, ck, false, false);
        const float bbq = bq[N], bbk = bk[N];
#pragma unroll
        for (int r = 0; r < 8; ++r) {
            const int row = node0 + r + (lo ? 0 : 8);
            float zq = cq[r] + bbq;
            float zk = ck[r] + bbk;
            Qo[row * C_HID + N] = zq > 0.f ? 1.f + zq : __expf(zq);  // 1+elu
            Ko[row * C_HID + N] = zk > 0.f ? 1.f + zk : __expf(zk);
        }
    }

    // -------- V = x@Wv + bv (one 16-wide N tile) --------
    {
        const int N  = m;
        const int kb = lo ? 0 : 16;
        h16 bvf;
#pragma unroll
        for (int h = 0; h < 16; ++h)
            bvf[h] = (_Float16)Wv[(kb + h) * N_CLS + N];
        f8 cv;
#pragma unroll
        for (int i = 0; i < 8; ++i) cv[i] = 0.f;
        cv = __builtin_amdgcn_wmma_f32_16x16x32_f16(false, xa, false, bvf, (short)0, cv, false, false);
        const float bbv = bv[N];
#pragma unroll
        for (int r = 0; r < 8; ++r) {
            const int row = node0 + r + (lo ? 0 : 8);
            Vo[row * N_CLS + N] = cv[r] + bbv;
        }
    }
}

// ---------------------------------------------------------------------------
// Kernel 2: M0[n,i,j] = K[n,i]*V[n,j]; hidden init = hopwise[0]*V
// ---------------------------------------------------------------------------
__global__ void init_M(const float* __restrict__ K, const float* __restrict__ V,
                       const float* __restrict__ hopwise,
                       float* __restrict__ M, float* __restrict__ out)
{
    const int t = blockIdx.x * blockDim.x + threadIdx.x;
    if (t >= NNODES * C_HID * N_CLS) return;
    const int j    = t & (N_CLS - 1);
    const int i    = (t >> 4) & (C_HID - 1);
    const int node = t >> 9;
    const float v  = V[node * N_CLS + j];
    M[t] = K[node * C_HID + i] * v;
    if (i == 0) out[node * N_CLS + j] = hopwise[0] * v;
}

__global__ void zero_f32(float* __restrict__ p, int n)
{
    const int t = blockIdx.x * blockDim.x + threadIdx.x;
    if (t < n) p[t] = 0.f;
}

// ---------------------------------------------------------------------------
// Kernel 3: edge scatter of M (128 threads/edge, float4 + hw f32 atomics)
// ---------------------------------------------------------------------------
__global__ __launch_bounds__(256)
void scatter_M(const int* __restrict__ row, const int* __restrict__ col,
               const float* __restrict__ Min, float* __restrict__ Mout)
{
    const long long t = (long long)blockIdx.x * blockDim.x + threadIdx.x;
    if (t >= (long long)NEDGES * 128) return;
    const int e = (int)(t >> 7);
    const int q = (int)(t & 127);
    const int r = row[e], c = col[e];
    const float4 v = *(const float4*)(Min + (size_t)r * 512 + q * 4);
    float* dst = Mout + (size_t)c * 512 + q * 4;
    unsafeAtomicAdd(dst + 0, v.x);
    unsafeAtomicAdd(dst + 1, v.y);
    unsafeAtomicAdd(dst + 2, v.z);
    unsafeAtomicAdd(dst + 3, v.w);
}

__global__ __launch_bounds__(256)
void scatter_K(const int* __restrict__ row, const int* __restrict__ col,
               const float* __restrict__ Kin, float* __restrict__ Kout)
{
    const long long t = (long long)blockIdx.x * blockDim.x + threadIdx.x;
    if (t >= (long long)NEDGES * 8) return;
    const int e = (int)(t >> 3);
    const int q = (int)(t & 7);
    const int r = row[e], c = col[e];
    const float4 v = *(const float4*)(Kin + (size_t)r * C_HID + q * 4);
    float* dst = Kout + (size_t)c * C_HID + q * 4;
    unsafeAtomicAdd(dst + 0, v.x);
    unsafeAtomicAdd(dst + 1, v.y);
    unsafeAtomicAdd(dst + 2, v.z);
    unsafeAtomicAdd(dst + 3, v.w);
}

// ---------------------------------------------------------------------------
// Kernel 4: H = Q.M, C = Q.K + cst, hidden += hopwise[hop]*(H/C)
// ---------------------------------------------------------------------------
__global__ void hop_combine(const float* __restrict__ Q, const float* __restrict__ K,
                            const float* __restrict__ M,
                            const float* __restrict__ hopwise, int hopidx,
                            float* __restrict__ out)
{
    const int t = blockIdx.x * blockDim.x + threadIdx.x;
    if (t >= NNODES * N_CLS) return;
    const int j    = t & (N_CLS - 1);
    const int node = t >> 4;
    const float* q = Q + node * C_HID;
    const float* k = K + node * C_HID;
    const float* m = M + (size_t)node * 512 + j;
    float h = 0.f, c = CSTF;
#pragma unroll
    for (int i = 0; i < C_HID; ++i) {
        const float qi = q[i];
        h += qi * m[i * N_CLS];
        c += qi * k[i];
    }
    out[t] += hopwise[hopidx] * (h / c);
}

// ---------------------------------------------------------------------------
extern "C" void kernel_launch(void* const* d_in, const int* in_sizes, int n_in,
                              void* d_out, int out_size, void* d_ws, size_t ws_size,
                              hipStream_t stream)
{
    const float* feat = (const float*)d_in[0];
    const int*   edge = (const int*)d_in[1];
    const int*   erow = edge;
    const int*   ecol = edge + NEDGES;
    const float* Win = (const float*)d_in[2];
    const float* bin = (const float*)d_in[3];
    const float* Wq  = (const float*)d_in[4];
    const float* bq  = (const float*)d_in[5];
    const float* Wk  = (const float*)d_in[6];
    const float* bk  = (const float*)d_in[7];
    const float* Wv  = (const float*)d_in[8];
    const float* bv  = (const float*)d_in[9];
    const float* hopwise = (const float*)d_in[10];
    float* out = (float*)d_out;

    float* ws = (float*)d_ws;
    float* Q  = ws;
    float* Ka = Q  + (size_t)NNODES * C_HID;
    float* Kb = Ka + (size_t)NNODES * C_HID;
    float* V  = Kb + (size_t)NNODES * C_HID;
    float* Ma = V  + (size_t)NNODES * N_CLS;
    float* Mb = Ma + (size_t)NNODES * C_HID * N_CLS;

    // 1) node transforms (WMMA): Q, K0, V
    node_transform<<<(NNODES / 16 + 7) / 8, 256, 0, stream>>>(
        feat, Win, bin, Wq, bq, Wk, bk, Wv, bv, Q, Ka, V);

    // 2) M0 = K0 (x) V ; hidden = hopwise[0]*V
    const int totM = NNODES * C_HID * N_CLS;
    init_M<<<(totM + 255) / 256, 256, 0, stream>>>(Ka, V, hopwise, Ma, out);

    float *Mi = Ma, *Mo = Mb, *Ki = Ka, *Ko = Kb;
    for (int hop = 0; hop < K_HOPS; ++hop) {
        zero_f32<<<(totM + 255) / 256, 256, 0, stream>>>(Mo, totM);
        zero_f32<<<(NNODES * C_HID + 255) / 256, 256, 0, stream>>>(Ko, NNODES * C_HID);

        const long long wM = (long long)NEDGES * 128;
        scatter_M<<<(unsigned)((wM + 255) / 256), 256, 0, stream>>>(erow, ecol, Mi, Mo);
        const long long wK = (long long)NEDGES * 8;
        scatter_K<<<(unsigned)((wK + 255) / 256), 256, 0, stream>>>(erow, ecol, Ki, Ko);

        hop_combine<<<(NNODES * N_CLS + 255) / 256, 256, 0, stream>>>(
            Q, Ko, Mo, hopwise, hop + 1, out);

        float* tm = Mi; Mi = Mo; Mo = tm;
        float* tk = Ki; Ki = Ko; Ko = tk;
    }
}
